// AttentionWin_69569880260873
// MI455X (gfx1250) — matrix operand order repaired and checked by reading, not compiled, not run on hardware
//
#include <hip/hip_runtime.h>
#include <hip/hip_bf16.h>

typedef __attribute__((ext_vector_type(16))) _Float16 v16h;
typedef __attribute__((ext_vector_type(8)))  _Float16 v8h;
typedef __attribute__((ext_vector_type(8)))  float    v8f;

// ---------------------------------------------------------------------------
// WMMA helper: D = A(16x32 f16) x B(32x16 f16) + C(16x16 f32)
// ---------------------------------------------------------------------------
__device__ __forceinline__ v8f wmma_f16(v16h a, v16h b, v8f c) {
  return __builtin_amdgcn_wmma_f32_16x16x32_f16(
      false, a, false, b, (short)0, c, false, false);
}

__device__ __forceinline__ v16h ld_frag_pair(const _Float16* p0, const _Float16* p1) {
  v8h lo = *(const v8h*)p0;
  v8h hi = *(const v8h*)p1;
  v16h r;
#pragma unroll
  for (int i = 0; i < 8; ++i) { r[i] = lo[i]; r[i + 8] = hi[i]; }
  return r;
}
__device__ __forceinline__ v16h ld_frag_contig(const _Float16* p) {
  return ld_frag_pair(p, p + 8);
}

// ---------------------------------------------------------------------------
// gfx1250 async copy global -> LDS (16B per lane), tracked by ASYNCcnt.
// LDS offset = low 32 bits of the generic pointer (flat LDS aperture maps
// addr[31:0] to the LDS offset per CDNA5 ISA 10.2).
// ---------------------------------------------------------------------------
__device__ __forceinline__ void async_copy_b128(const _Float16* g, _Float16* l) {
  unsigned loff = (unsigned)(unsigned long long)l;
  unsigned long long ga = (unsigned long long)g;
  asm volatile("global_load_async_to_lds_b128 %0, %1, off"
               :: "v"(loff), "v"(ga) : "memory");
}
__device__ __forceinline__ void wait_asynccnt0() {
  asm volatile("s_wait_asynccnt 0x0" ::: "memory");
}

// ---------------------------------------------------------------------------
// Prep kernels: G_t = einsum(pqv,v), A1 = U1 @ G_t, W_eff = W + A1 @ U2^T (f16)
// ---------------------------------------------------------------------------
__global__ void prep_gt(const float* __restrict__ Gq, const float* __restrict__ U3q,
                        const float* __restrict__ Gp, const float* __restrict__ U3p,
                        const int* __restrict__ task, float* __restrict__ gt) {
  int t = threadIdx.x;               // 0..127
  int which = t >> 6;                // 0 = qkv, 1 = proj
  int pq = t & 63;
  const float* G  = which ? Gp  : Gq;
  const float* U3 = which ? U3p : U3q;
  int ti = task[0];
  float s = 0.f;
#pragma unroll
  for (int v = 0; v < 8; ++v) s += G[pq * 8 + v] * U3[ti * 8 + v];
  gt[t] = s;
}

__global__ void prep_a1(const float* __restrict__ U1q, const float* __restrict__ U1p,
                        const float* __restrict__ gt, float* __restrict__ a1) {
  int o = blockIdx.x * blockDim.x + threadIdx.x;   // 0..3071 (2304 qkv + 768 proj)
  if (o >= 3072) return;
  bool proj = (o >= 2304);
  const float* U1 = proj ? (U1p + (size_t)(o - 2304) * 8) : (U1q + (size_t)o * 8);
  const float* g  = gt + (proj ? 64 : 0);
#pragma unroll
  for (int q = 0; q < 8; ++q) {
    float s = 0.f;
#pragma unroll
    for (int p = 0; p < 8; ++p) s += U1[p] * g[p * 8 + q];
    a1[(size_t)o * 8 + q] = s;
  }
}

__global__ void prep_weff(const float* __restrict__ Wq, const float* __restrict__ U2q,
                          const float* __restrict__ Wp, const float* __restrict__ U2p,
                          const float* __restrict__ a1,
                          _Float16* __restrict__ wq16, _Float16* __restrict__ wp16) {
  size_t idx = (size_t)blockIdx.x * blockDim.x + threadIdx.x;
  const size_t NQ = (size_t)2304 * 768;
  const size_t NP = (size_t)768 * 768;
  if (idx < NQ) {
    int o = (int)(idx / 768), i = (int)(idx % 768);
    const float* ar = a1 + (size_t)o * 8;
    const float* u2 = U2q + (size_t)i * 8;
    float s = Wq[idx];
#pragma unroll
    for (int q = 0; q < 8; ++q) s += ar[q] * u2[q];
    wq16[idx] = (_Float16)s;
  } else if (idx < NQ + NP) {
    size_t j = idx - NQ;
    int o = (int)(j / 768), i = (int)(j % 768);
    const float* ar = a1 + (size_t)(2304 + o) * 8;
    const float* u2 = U2p + (size_t)i * 8;
    float s = Wp[j];
#pragma unroll
    for (int q = 0; q < 8; ++q) s += ar[q] * u2[q];
    wp16[j] = (_Float16)s;
  }
}

// ---------------------------------------------------------------------------
// WMMA GEMM: C[M,N] = A[M,K](f32) * B[N,K](f16, K-contig)^T + bias
// 256 threads / 8 waves; block tile 64(M) x 128(N), K step 32.
// Each wave: 32x32 (2x2 wmma). Double-buffered LDS: B via async-to-LDS,
// A via register prefetch (f32 -> f16 convert). One barrier per K step.
// ---------------------------------------------------------------------------
__global__ __launch_bounds__(256)
void gemm_wmma_kernel(const float* __restrict__ A, const _Float16* __restrict__ Bw,
                      const float* __restrict__ bias,
                      float* __restrict__ Cf32, _Float16* __restrict__ Cf16,
                      int M, int N, int K, float qscale, int qcols) {
  __shared__ __align__(16) _Float16 ldsA[2][64 * 32];
  __shared__ __align__(16) _Float16 ldsB[2][128 * 32];

  const int tid  = threadIdx.x;
  const int lane = tid & 31, wave = tid >> 5;
  const int half = lane >> 4, ln = lane & 15;
  const int wm = wave >> 2, wn = wave & 3;           // 2 x 4 wave grid
  const int m0 = blockIdx.y * 64, n0 = blockIdx.x * 128;

  const int ar = tid >> 2, ac = (tid & 3) * 8;       // A staging: 8 f32 / thread
  const int br = tid >> 1, bc = (tid & 1) * 16;      // B staging: 16 f16 / thread
  const int nsteps = K >> 5;

  const float*    Ab = A  + (size_t)(m0 + ar) * K + ac;
  const _Float16* Bb = Bw + (size_t)(n0 + br) * K + bc;

  v8f acc[2][2] = {};

  // Prologue: stage tile 0 into buffer 0.
  async_copy_b128(Bb,     &ldsB[0][br * 32 + bc]);
  async_copy_b128(Bb + 8, &ldsB[0][br * 32 + bc + 8]);
  {
    float4 a0 = *(const float4*)(Ab);
    float4 a1 = *(const float4*)(Ab + 4);
    _Float16* d = &ldsA[0][ar * 32 + ac];
    d[0] = (_Float16)a0.x; d[1] = (_Float16)a0.y; d[2] = (_Float16)a0.z; d[3] = (_Float16)a0.w;
    d[4] = (_Float16)a1.x; d[5] = (_Float16)a1.y; d[6] = (_Float16)a1.z; d[7] = (_Float16)a1.w;
  }
  wait_asynccnt0();
  __syncthreads();

  for (int ks = 0; ks < nsteps; ++ks) {
    const int cur = ks & 1, nxt = cur ^ 1;
    const bool more = (ks + 1) < nsteps;
    float4 pa0, pa1;
    if (more) {
      const int kk = (ks + 1) * 32;
      // Async copy of next B tile overlaps this step's wmmas.
      async_copy_b128(Bb + kk,     &ldsB[nxt][br * 32 + bc]);
      async_copy_b128(Bb + kk + 8, &ldsB[nxt][br * 32 + bc + 8]);
      pa0 = *(const float4*)(Ab + kk);
      pa1 = *(const float4*)(Ab + kk + 4);
    }

    const _Float16* lA = ldsA[cur];
    const _Float16* lB = ldsB[cur];
    v16h aF[2], bF[2];
#pragma unroll
    for (int mi = 0; mi < 2; ++mi) {
      const _Float16* ap = lA + (wm * 32 + mi * 16 + ln) * 32;
      aF[mi] = ld_frag_pair(ap + 8 * half, ap + 16 + 8 * half);
    }
#pragma unroll
    for (int ni = 0; ni < 2; ++ni)
      bF[ni] = ld_frag_pair(lB + (wn * 32 + ni * 16 + ln) * 32 + 8 * half, lB + (wn * 32 + ni * 16 + ln) * 32 + 16 + 8 * half);
#pragma unroll
    for (int mi = 0; mi < 2; ++mi)
#pragma unroll
      for (int ni = 0; ni < 2; ++ni)
        acc[mi][ni] = wmma_f16(aF[mi], bF[ni], acc[mi][ni]);

    if (more) {
      _Float16* d = &ldsA[nxt][ar * 32 + ac];
      d[0] = (_Float16)pa0.x; d[1] = (_Float16)pa0.y; d[2] = (_Float16)pa0.z; d[3] = (_Float16)pa0.w;
      d[4] = (_Float16)pa1.x; d[5] = (_Float16)pa1.y; d[6] = (_Float16)pa1.z; d[7] = (_Float16)pa1.w;
      wait_asynccnt0();
      __syncthreads();
    }
  }

  // Epilogue: bias (+ optional q-scale), f16 or f32 store.
#pragma unroll
  for (int ni = 0; ni < 2; ++ni) {
    const int col = n0 + wn * 32 + ni * 16 + ln;
    const float b = bias[col];
    const float sc = (col < qcols) ? qscale : 1.0f;
#pragma unroll
    for (int mi = 0; mi < 2; ++mi) {
#pragma unroll
      for (int e = 0; e < 8; ++e) {
        int row = m0 + wm * 32 + mi * 16 + 8 * half + e;
        float v = (acc[mi][ni][e] + b) * sc;
        if (Cf16) Cf16[(size_t)row * N + col] = (_Float16)v;
        else      Cf32[(size_t)row * N + col] = v;
      }
    }
  }
}

// ---------------------------------------------------------------------------
// Flash attention: qkv f16 [8192, 2304] (q pre-scaled), out f32 [8192, 768].
// Block = (b, h, 128-query tile); wave = 16 queries; kv step = 32 keys.
// Double-buffered: K via async-to-LDS, V via register prefetch + transpose.
// ---------------------------------------------------------------------------
__global__ __launch_bounds__(256)
void attn_kernel(const _Float16* __restrict__ qkv, float* __restrict__ attnout) {
  __shared__ __align__(16) _Float16 ldsK[2][32 * 64];      // [key][d] row-major
  __shared__ __align__(16) _Float16 ldsVt[2][64 * 32];     // [d][key] transposed
  __shared__ __align__(16) _Float16 ldsP[8 * 16 * 32];     // per-wave P tiles

  const int tid  = threadIdx.x;
  const int lane = tid & 31, wave = tid >> 5;
  const int half = lane >> 4, ln = lane & 15;
  const int b = blockIdx.z, h = blockIdx.y;
  const int q0 = blockIdx.x * 128 + wave * 16;

  const _Float16* Qb = qkv + (size_t)b * 1024 * 2304 + h * 64;
  const _Float16* Kb = Qb + 768;
  const _Float16* Vb = Qb + 1536;

  // Preload Q fragments (A-matrix, two K=32 chunks over hd=64).
  const _Float16* qp = Qb + (size_t)(q0 + ln) * 2304;
  v16h aQ[2];
#pragma unroll
  for (int c = 0; c < 2; ++c)
    aQ[c] = ld_frag_pair(qp + c * 32 + 8 * half, qp + c * 32 + 16 + 8 * half);

  float m_i[8], l_i[8];
  v8f outAcc[4] = {};
#pragma unroll
  for (int e = 0; e < 8; ++e) { m_i[e] = -1e30f; l_i[e] = 0.f; }

  const int skey = tid >> 3;            // 0..31 staging key
  const int sc8  = (tid & 7) * 8;       // 0..56 staging d chunk
  const _Float16* Kst = Kb + (size_t)skey * 2304 + sc8;
  const _Float16* Vst = Vb + (size_t)skey * 2304 + sc8;

  // Prologue: stage kv tile 0 into buffer 0.
  async_copy_b128(Kst, &ldsK[0][skey * 64 + sc8]);
  {
    v8h vv = *(const v8h*)(Vst);
#pragma unroll
    for (int j = 0; j < 8; ++j) ldsVt[0][(sc8 + j) * 32 + skey] = vv[j];
  }
  wait_asynccnt0();
  __syncthreads();

  for (int it = 0; it < 32; ++it) {
    const int cur = it & 1, nxt = cur ^ 1;
    const bool more = (it + 1) < 32;
    v8h pv;
    if (more) {
      const size_t koff = (size_t)(it + 1) * 32 * 2304;
      async_copy_b128(Kst + koff, &ldsK[nxt][skey * 64 + sc8]);
      pv = *(const v8h*)(Vst + koff);
    }

    // Scores: two 16x16 tiles (keys 0-15, 16-31), K-dim = hd = 64.
    v8f accS[2];
#pragma unroll
    for (int s = 0; s < 2; ++s) {
      v8f a = {};
      const int keyl = s * 16 + ln;
#pragma unroll
      for (int c = 0; c < 2; ++c) {
        v16h bK = ld_frag_pair(&ldsK[cur][keyl * 64 + c * 32 + 8 * half], &ldsK[cur][keyl * 64 + c * 32 + 16 + 8 * half]);
        a = wmma_f16(aQ[c], bK, a);
      }
      accS[s] = a;
    }

    // Online softmax: per-row (e) stats reduced over the 16 lanes of a row.
    _Float16* pw = &ldsP[wave * 512];
#pragma unroll
    for (int e = 0; e < 8; ++e) {
      float cm = fmaxf(accS[0][e], accS[1][e]);
      cm = fmaxf(cm, __shfl_xor(cm, 1, 32));
      cm = fmaxf(cm, __shfl_xor(cm, 2, 32));
      cm = fmaxf(cm, __shfl_xor(cm, 4, 32));
      cm = fmaxf(cm, __shfl_xor(cm, 8, 32));
      float mn = fmaxf(m_i[e], cm);
      float al = __expf(m_i[e] - mn);
      float p0 = __expf(accS[0][e] - mn);
      float p1 = __expf(accS[1][e] - mn);
      float rs = p0 + p1;
      rs += __shfl_xor(rs, 1, 32);
      rs += __shfl_xor(rs, 2, 32);
      rs += __shfl_xor(rs, 4, 32);
      rs += __shfl_xor(rs, 8, 32);
      l_i[e] = l_i[e] * al + rs;
      m_i[e] = mn;
#pragma unroll
      for (int nt = 0; nt < 4; ++nt) outAcc[nt][e] *= al;
      pw[(8 * half + e) * 32 + ln]      = (_Float16)p0;
      pw[(8 * half + e) * 32 + 16 + ln] = (_Float16)p1;
    }

    // P (16x32) as A fragment; V from transposed LDS as B fragments.
    v16h aP = ld_frag_pair(&pw[ln * 32 + 8 * half], &pw[ln * 32 + 16 + 8 * half]);
#pragma unroll
    for (int nt = 0; nt < 4; ++nt) {
      v16h bV = ld_frag_pair(&ldsVt[cur][(nt * 16 + ln) * 32 + 8 * half], &ldsVt[cur][(nt * 16 + ln) * 32 + 16 + 8 * half]);
      outAcc[nt] = wmma_f16(aP, bV, outAcc[nt]);
    }

    if (more) {
#pragma unroll
      for (int j = 0; j < 8; ++j) ldsVt[nxt][(sc8 + j) * 32 + skey] = pv[j];
      wait_asynccnt0();
      __syncthreads();
    }
  }

  // Epilogue: normalize, scatter to [token, h*64 + d] (f32).
#pragma unroll
  for (int e = 0; e < 8; ++e) {
    float inv = 1.0f / l_i[e];
    int q = q0 + 8 * half + e;
    float* op = attnout + (size_t)(b * 1024 + q) * 768 + h * 64;
#pragma unroll
    for (int nt = 0; nt < 4; ++nt) op[nt * 16 + ln] = outAcc[nt][e] * inv;
  }
}

// ---------------------------------------------------------------------------
// Host launcher
// ---------------------------------------------------------------------------
extern "C" void kernel_launch(void* const* d_in, const int* in_sizes, int n_in,
                              void* d_out, int out_size, void* d_ws, size_t ws_size,
                              hipStream_t stream) {
  const float* x      = (const float*)d_in[0];
  const float* qkv_w  = (const float*)d_in[1];
  const float* qkv_b  = (const float*)d_in[2];
  const float* U1q    = (const float*)d_in[3];
  const float* U2q    = (const float*)d_in[4];
  const float* U3q    = (const float*)d_in[5];
  const float* Gq     = (const float*)d_in[6];
  const float* proj_w = (const float*)d_in[7];
  const float* proj_b = (const float*)d_in[8];
  const float* U1p    = (const float*)d_in[9];
  const float* U2p    = (const float*)d_in[10];
  const float* U3p    = (const float*)d_in[11];
  const float* Gp     = (const float*)d_in[12];
  const int*   task   = (const int*)d_in[13];
  float* out = (float*)d_out;

  char* ws = (char*)d_ws;
  float*    gt      = (float*)(ws + 0);            // 128 f32
  float*    a1      = (float*)(ws + 1024);         // 3072 x 8 f32
  _Float16* wq16    = (_Float16*)(ws + 99328);     // 2304 x 768 f16
  _Float16* wp16    = (_Float16*)(ws + 3638272);   // 768 x 768 f16
  _Float16* qkvbuf  = (_Float16*)(ws + 4817920);   // 8192 x 2304 f16
  float*    attnbuf = (float*)(ws + 42566656);     // 8192 x 768 f32

  prep_gt<<<1, 128, 0, stream>>>(Gq, U3q, Gp, U3p, task, gt);
  prep_a1<<<12, 256, 0, stream>>>(U1q, U1p, gt, a1);
  prep_weff<<<9216, 256, 0, stream>>>(qkv_w, U2q, proj_w, U2p, a1, wq16, wp16);

  // QKV GEMM: [8192,768] x [2304,768]^T + b, q columns pre-scaled by 1/8.
  dim3 g1(2304 / 128, 8192 / 64);
  gemm_wmma_kernel<<<g1, 256, 0, stream>>>(x, wq16, qkv_b, nullptr, qkvbuf,
                                           8192, 2304, 768, 0.125f, 768);
  // Attention: 8 query tiles x 12 heads x 8 batches.
  dim3 g2(8, 12, 8);
  attn_kernel<<<g2, 256, 0, stream>>>(qkvbuf, attnbuf);

  // Proj GEMM: [8192,768] x [768,768]^T + b -> f32 output.
  dim3 g3(768 / 128, 8192 / 64);
  gemm_wmma_kernel<<<g3, 256, 0, stream>>>(attnbuf, wp16, proj_b, out, nullptr,
                                           8192, 768, 768, 1.0f, 0);
}
